// FlaxLlamaAttention_24197845745708
// MI455X (gfx1250) — compile-verified, matrix operand order checked by reading, not run on hardware
//
#include <hip/hip_runtime.h>

typedef __bf16 bf16_t;
typedef __attribute__((ext_vector_type(8)))  __bf16 v8bf;
typedef __attribute__((ext_vector_type(16))) __bf16 v16bf;
typedef __attribute__((ext_vector_type(8)))  float  v8f;
typedef long long i64;

#define WMMA_BF16(a, b, c) \
  __builtin_amdgcn_wmma_f32_16x16x32_bf16(false, (a), false, (b), (short)0, (c), false, false)

// Load one 16x32 WMMA operand (A as MxK, or B as NxK i.e. transposed) from a
// row-major [row][K] bf16 buffer. Per ISA 7.12.2 (16-bit A 16x32 layout):
// lane L holds row L%16; elements 0..7 <-> K = halfoff..halfoff+7,
// elements 8..15 <-> K = halfoff+16..halfoff+23, halfoff = (L>=16)*8.
// Both 8-element groups are contiguous -> two b128 loads.
__device__ __forceinline__ v16bf load_op16(const bf16_t* __restrict__ base,
                                           i64 row, int ld, int kbase, int lane) {
  const bf16_t* p = base + row * (i64)ld + kbase + ((lane & 16) ? 8 : 0);
  v8bf lo = *(const v8bf*)(p);
  v8bf hi = *(const v8bf*)(p + 16);
  return __builtin_shufflevector(lo, hi, 0, 1, 2, 3, 4, 5, 6, 7,
                                 8, 9, 10, 11, 12, 13, 14, 15);
}

__device__ __forceinline__ v8f v8f_zero() {
  v8f z = {0.f, 0.f, 0.f, 0.f, 0.f, 0.f, 0.f, 0.f};
  return z;
}

// ---------------------------------------------------------------------------
// Wt[n][k] = (bf16) W[k][n]   (weights pre-transposed so B-operand is row-major N x K)
__global__ void wt_transpose_bf16(const float* __restrict__ W, bf16_t* __restrict__ Wt,
                                  int N, int K) {
  i64 idx = (i64)blockIdx.x * blockDim.x + threadIdx.x;
  i64 total = (i64)N * K;
  if (idx >= total) return;
  int k = (int)(idx % K);
  int n = (int)(idx / K);
  Wt[(i64)n * K + k] = (bf16_t)W[(i64)k * N + n];
}

__global__ void f32_to_bf16(const float* __restrict__ in, bf16_t* __restrict__ out, i64 n) {
  i64 idx = (i64)blockIdx.x * blockDim.x + threadIdx.x;
  if (idx < n) out[idx] = (bf16_t)in[idx];
}

// ---------------------------------------------------------------------------
// C[M][N] = A[M][K] (bf16, row-major) x Bt[N][K] (bf16, B transposed)
// mode 0: write bf16 head-major  [h][m][d]  (h = n>>7, d = n&127, stride uses M==S)
// mode 1: write bf16 head-major transposed [h][d][m]   (for V)
// mode 2: write f32 row-major    [m][n]
// Block = 128 threads = 4 waves; each wave owns a 32(M) x 64(N) tile.
// k-loop unrolled by 2 with ping-pong operand sets: no register rotation,
// loads of one set overlap the WMMAs consuming the other set.
// Requires K % 64 == 0 and K >= 128 (holds: K = 2048).
__global__ void __launch_bounds__(128)
gemm_bf16_wmma(const bf16_t* __restrict__ A, const bf16_t* __restrict__ Bt,
               void* __restrict__ Cout, int M, int N, int K, int mode) {
  int lane = threadIdx.x & 31;
  int wave = threadIdx.x >> 5;
  int m0 = blockIdx.y * 128 + wave * 32;
  int n0 = blockIdx.x * 64;
  int nl = lane & 15;

  v8f acc[2][4];
#pragma unroll
  for (int i = 0; i < 2; ++i)
#pragma unroll
    for (int c = 0; c < 4; ++c) acc[i][c] = v8f_zero();

  // ping-pong operand sets
  v16bf a0A, a1A, bA[4];
  v16bf a0B, a1B, bB[4];

  // preload set A @ k = 0
  a0A = load_op16(A, m0 + nl, K, 0, lane);
  a1A = load_op16(A, m0 + 16 + nl, K, 0, lane);
#pragma unroll
  for (int c = 0; c < 4; ++c) bA[c] = load_op16(Bt, n0 + c * 16 + nl, K, 0, lane);

  for (int k = 0; k < K - 64; k += 64) {
    // load set B @ k+32 while consuming set A @ k
    a0B = load_op16(A, m0 + nl, K, k + 32, lane);
    a1B = load_op16(A, m0 + 16 + nl, K, k + 32, lane);
#pragma unroll
    for (int c = 0; c < 4; ++c) bB[c] = load_op16(Bt, n0 + c * 16 + nl, K, k + 32, lane);
#pragma unroll
    for (int c = 0; c < 4; ++c) {
      acc[0][c] = WMMA_BF16(a0A, bA[c], acc[0][c]);
      acc[1][c] = WMMA_BF16(a1A, bA[c], acc[1][c]);
    }
    // load set A @ k+64 while consuming set B @ k+32
    a0A = load_op16(A, m0 + nl, K, k + 64, lane);
    a1A = load_op16(A, m0 + 16 + nl, K, k + 64, lane);
#pragma unroll
    for (int c = 0; c < 4; ++c) bA[c] = load_op16(Bt, n0 + c * 16 + nl, K, k + 64, lane);
#pragma unroll
    for (int c = 0; c < 4; ++c) {
      acc[0][c] = WMMA_BF16(a0B, bB[c], acc[0][c]);
      acc[1][c] = WMMA_BF16(a1B, bB[c], acc[1][c]);
    }
  }

  // tail: set A holds block K-64; load final block K-32 into set B and drain
  a0B = load_op16(A, m0 + nl, K, K - 32, lane);
  a1B = load_op16(A, m0 + 16 + nl, K, K - 32, lane);
#pragma unroll
  for (int c = 0; c < 4; ++c) bB[c] = load_op16(Bt, n0 + c * 16 + nl, K, K - 32, lane);
#pragma unroll
  for (int c = 0; c < 4; ++c) {
    acc[0][c] = WMMA_BF16(a0A, bA[c], acc[0][c]);
    acc[1][c] = WMMA_BF16(a1A, bA[c], acc[1][c]);
  }
#pragma unroll
  for (int c = 0; c < 4; ++c) {
    acc[0][c] = WMMA_BF16(a0B, bB[c], acc[0][c]);
    acc[1][c] = WMMA_BF16(a1B, bB[c], acc[1][c]);
  }

  int half8 = (lane & 16) ? 8 : 0;
#pragma unroll
  for (int i = 0; i < 2; ++i) {
#pragma unroll
    for (int c = 0; c < 4; ++c) {
#pragma unroll
      for (int r = 0; r < 8; ++r) {
        i64 m = m0 + i * 16 + r + half8;   // C/D layout: VGPR r -> row r (+8 hi lanes)
        int n = n0 + c * 16 + nl;          // lane -> column
        float v = acc[i][c][r];
        if (mode == 0) {
          int h = n >> 7, d = n & 127;
          ((bf16_t*)Cout)[((i64)h * M + m) * 128 + d] = (bf16_t)v;
        } else if (mode == 1) {
          int h = n >> 7, d = n & 127;
          ((bf16_t*)Cout)[((i64)h * 128 + d) * M + m] = (bf16_t)v;
        } else {
          ((float*)Cout)[m * (i64)N + n] = v;
        }
      }
    }
  }
}

// ---------------------------------------------------------------------------
// In-place RoPE on head-major Q/K [H][S][128]; Q additionally scaled by 1/sqrt(128).
__global__ void rope_inplace(bf16_t* __restrict__ Q, bf16_t* __restrict__ K,
                             const int* __restrict__ pos_ids, int S) {
  i64 idx = (i64)blockIdx.x * blockDim.x + threadIdx.x;
  i64 total = (i64)16 * S * 64;
  if (idx >= total) return;
  int d = (int)(idx & 63);
  int s = (int)((idx >> 6) % S);
  int h = (int)((idx >> 6) / S);

  float pos = (float)pos_ids[s];
  float inv = __powf(10000.f, -(float)d * (1.f / 64.f));
  float ang = pos * inv;
  float sn, cs;
  __sincosf(ang, &sn, &cs);

  bf16_t* q = Q + ((i64)h * S + s) * 128;
  bf16_t* kk = K + ((i64)h * S + s) * 128;

  const float scale = 0.08838834764831845f;  // 1/sqrt(128)
  float q0 = (float)q[d], q1 = (float)q[d + 64];
  q[d]      = (bf16_t)((q0 * cs - q1 * sn) * scale);
  q[d + 64] = (bf16_t)((q1 * cs + q0 * sn) * scale);

  float k0 = (float)kk[d], k1 = (float)kk[d + 64];
  kk[d]      = (bf16_t)(k0 * cs - k1 * sn);
  kk[d + 64] = (bf16_t)(k1 * cs + k0 * sn);
}

// ---------------------------------------------------------------------------
// Flash attention: one wave handles 16 query rows of one head.
// Qbf/Kbf: [H][S][128] bf16 (Q pre-scaled & rotated), Vt: [H][128][S] bf16.
// Writes attention output bf16 row-major [S][2048] (head-interleaved cols).
__global__ void __launch_bounds__(32)
flash_attn_wmma(const bf16_t* __restrict__ Qbf, const bf16_t* __restrict__ Kbf,
                const bf16_t* __restrict__ Vt, const int* __restrict__ amask,
                bf16_t* __restrict__ Obf, int S) {
  const int D = 128;
  int lane = threadIdx.x & 31;
  int qb = blockIdx.x;          // 16-query block
  int h  = blockIdx.y;
  int nl = lane & 15;
  int half8 = (lane & 16) ? 8 : 0;

  const bf16_t* Qh = Qbf + (i64)h * S * D;
  const bf16_t* Kh = Kbf + (i64)h * S * D;
  const bf16_t* Vh = Vt  + (i64)h * D * S;

  v16bf qa[4];
#pragma unroll
  for (int c = 0; c < 4; ++c) qa[c] = load_op16(Qh, qb * 16 + nl, D, c * 32, lane);

  v8f o[8];
#pragma unroll
  for (int c = 0; c < 8; ++c) o[c] = v8f_zero();
  float mr[8], lr[8];
#pragma unroll
  for (int r = 0; r < 8; ++r) { mr[r] = -3.0e38f; lr[r] = 0.f; }

  __shared__ __align__(16) bf16_t pl[16 * 32];   // P tile staging (C-layout -> A-layout)

  const float NEG = -3.0e38f;
  int q_hi = qb * 16 + 15;
  for (int kb = 0; kb <= q_hi; kb += 32) {
    // batch all 8 K-operand loads, then chain the score WMMAs
    v16bf k0[4], k1[4];
#pragma unroll
    for (int c = 0; c < 4; ++c) k0[c] = load_op16(Kh, kb + nl, D, c * 32, lane);
#pragma unroll
    for (int c = 0; c < 4; ++c) k1[c] = load_op16(Kh, kb + 16 + nl, D, c * 32, lane);

    v8f s0 = v8f_zero(), s1 = v8f_zero();
#pragma unroll
    for (int c = 0; c < 4; ++c) {
      s0 = WMMA_BF16(qa[c], k0[c], s0);
      s1 = WMMA_BF16(qa[c], k1[c], s1);
    }

    // prefetch next key block (K rows are 256B: two cachelines per row)
    if (kb + 32 <= q_hi) {
      const char* pk = (const char*)(Kh + (i64)(kb + 32 + lane) * D);
      __builtin_prefetch(pk, 0, 3);
      __builtin_prefetch(pk + 128, 0, 3);
    }

    int key0 = kb + nl, key1 = key0 + 16;
    float ok0 = (amask[key0] > 0) ? 0.f : NEG;
    float ok1 = (amask[key1] > 0) ? 0.f : NEG;

    float e0[8], e1[8], alpha[8];
#pragma unroll
    for (int r = 0; r < 8; ++r) {
      int qr = qb * 16 + r + half8;          // global query row for this accum row
      float x0 = (key0 <= qr) ? (s0[r] + ok0) : NEG;   // causal + padding mask
      float x1 = (key1 <= qr) ? (s1[r] + ok1) : NEG;
      float bm = fmaxf(x0, x1);
      bm = fmaxf(bm, __shfl_xor(bm, 1));
      bm = fmaxf(bm, __shfl_xor(bm, 2));
      bm = fmaxf(bm, __shfl_xor(bm, 4));
      bm = fmaxf(bm, __shfl_xor(bm, 8));     // row max across the 16-lane half
      float mnew = fmaxf(mr[r], bm);
      float a  = __expf(mr[r] - mnew);
      float p0 = __expf(x0 - mnew);
      float p1 = __expf(x1 - mnew);
      float rs = p0 + p1;
      rs += __shfl_xor(rs, 1);
      rs += __shfl_xor(rs, 2);
      rs += __shfl_xor(rs, 4);
      rs += __shfl_xor(rs, 8);
      lr[r] = lr[r] * a + rs;
      mr[r] = mnew;
      alpha[r] = a; e0[r] = p0; e1[r] = p1;
    }

#pragma unroll
    for (int c = 0; c < 8; ++c)
#pragma unroll
      for (int r = 0; r < 8; ++r) o[c][r] *= alpha[r];

    // stage P (16 rows x 32 keys) through LDS to get the A-operand layout
#pragma unroll
    for (int r = 0; r < 8; ++r) {
      pl[(r + half8) * 32 + nl]      = (bf16_t)e0[r];
      pl[(r + half8) * 32 + 16 + nl] = (bf16_t)e1[r];
    }
    __syncthreads();
    v16bf pop = load_op16(pl, nl, 32, 0, lane);

    // O += P x V : batch all 8 V-operand loads, then 8 independent WMMAs
    v16bf vb[8];
#pragma unroll
    for (int c = 0; c < 8; ++c) vb[c] = load_op16(Vh, c * 16 + nl, S, kb, lane);
#pragma unroll
    for (int c = 0; c < 8; ++c) o[c] = WMMA_BF16(pop, vb[c], o[c]);

    // prefetch next V columns (covers d = 0..127 across the 32 lanes)
    if (kb + 32 <= q_hi) {
#pragma unroll
      for (int j = 0; j < 4; ++j)
        __builtin_prefetch((const char*)(Vh + (i64)(lane * 4 + j) * S + kb + 32), 0, 3);
    }
    __syncthreads();
  }

#pragma unroll
  for (int r = 0; r < 8; ++r) lr[r] = 1.f / lr[r];
#pragma unroll
  for (int c = 0; c < 8; ++c)
#pragma unroll
    for (int r = 0; r < 8; ++r) {
      i64 m = qb * 16 + r + half8;
      int n = h * 128 + c * 16 + nl;
      Obf[m * 2048 + n] = (bf16_t)(o[c][r] * lr[r]);
    }
}

// ---------------------------------------------------------------------------
extern "C" void kernel_launch(void* const* d_in, const int* in_sizes, int n_in,
                              void* d_out, int out_size, void* d_ws, size_t ws_size,
                              hipStream_t stream) {
  (void)in_sizes; (void)n_in; (void)out_size; (void)ws_size;
  const float* X   = (const float*)d_in[0];
  const int*   am  = (const int*)d_in[1];
  const int*   pid = (const int*)d_in[2];
  const float* Wq  = (const float*)d_in[3];
  const float* Wk  = (const float*)d_in[4];
  const float* Wv  = (const float*)d_in[5];
  const float* Wo  = (const float*)d_in[6];

  const int S = 4096, E = 2048, H = 16;
  const i64 SE = (i64)S * E;   // 8M elems
  const i64 EE = (i64)E * E;   // 4M elems

  char* ws = (char*)d_ws;
  bf16_t* WtQ = (bf16_t*)ws; ws += EE * 2;
  bf16_t* WtK = (bf16_t*)ws; ws += EE * 2;
  bf16_t* WtV = (bf16_t*)ws; ws += EE * 2;
  bf16_t* WtO = (bf16_t*)ws; ws += EE * 2;
  bf16_t* Xbf = (bf16_t*)ws; ws += SE * 2;
  bf16_t* Qbf = (bf16_t*)ws; ws += SE * 2;   // [H][S][128]
  bf16_t* Kbf = (bf16_t*)ws; ws += SE * 2;   // [H][S][128]
  bf16_t* Vt  = (bf16_t*)ws; ws += SE * 2;   // [H][128][S]
  bf16_t* Obf = (bf16_t*)ws; ws += SE * 2;   // [S][E]

  const int tb = 256;
  wt_transpose_bf16<<<(unsigned)((EE + tb - 1) / tb), tb, 0, stream>>>(Wq, WtQ, E, E);
  wt_transpose_bf16<<<(unsigned)((EE + tb - 1) / tb), tb, 0, stream>>>(Wk, WtK, E, E);
  wt_transpose_bf16<<<(unsigned)((EE + tb - 1) / tb), tb, 0, stream>>>(Wv, WtV, E, E);
  wt_transpose_bf16<<<(unsigned)((EE + tb - 1) / tb), tb, 0, stream>>>(Wo, WtO, E, E);
  f32_to_bf16<<<(unsigned)((SE + tb - 1) / tb), tb, 0, stream>>>(X, Xbf, SE);

  dim3 gg(E / 64, S / 128);
  gemm_bf16_wmma<<<gg, 128, 0, stream>>>(Xbf, WtQ, Qbf, S, E, E, 0);
  gemm_bf16_wmma<<<gg, 128, 0, stream>>>(Xbf, WtK, Kbf, S, E, E, 0);
  gemm_bf16_wmma<<<gg, 128, 0, stream>>>(Xbf, WtV, Vt,  S, E, E, 1);

  i64 rope_total = (i64)H * S * 64;
  rope_inplace<<<(unsigned)((rope_total + tb - 1) / tb), tb, 0, stream>>>(Qbf, Kbf, pid, S);

  flash_attn_wmma<<<dim3(S / 16, H), 32, 0, stream>>>(Qbf, Kbf, Vt, am, Obf, S);

  gemm_bf16_wmma<<<gg, 128, 0, stream>>>(Obf, WtO, d_out, S, E, E, 2);
}
